// CoreRNNFW_27960237097125
// MI455X (gfx1250) — matrix operand order repaired and checked by reading, not compiled, not run on hardware
//
#include <hip/hip_runtime.h>

typedef __bf16 bf16;
typedef __attribute__((ext_vector_type(16))) __bf16 v16bf;
typedef __attribute__((ext_vector_type(8)))  __bf16 v8bf;
typedef __attribute__((ext_vector_type(8)))  float  v8f;

#define T_STEPS 16
#define BSZ     64
#define DG      512
#define DH      1024
#define DOUT    512
#define S_REF   3
#define LAMBDA  0.95f
#define ETA_C   0.5f
#define EPS_C   1e-6f
#define LN_EPS_C 1e-5f

// ---------------------------------------------------------------------------
// Elementwise helpers
// ---------------------------------------------------------------------------
__global__ void convert_f32_bf16_kernel(const float* __restrict__ src,
                                        bf16* __restrict__ dst, int n) {
  int i = blockIdx.x * blockDim.x + threadIdx.x;
  if (i < n) dst[i] = (bf16)src[i];
}

__global__ void zero_bf16_kernel(bf16* __restrict__ dst, int n) {
  int i = blockIdx.x * blockDim.x + threadIdx.x;
  if (i < n) dst[i] = (bf16)0.0f;
}

// ---------------------------------------------------------------------------
// WMMA GEMM:  C[64 x N] = A1[64 x K1] * B1^T + A2[64 x K2] * B2^T + bias
// B matrices given row-major as weight[n][k] (so B_tile[k][n] = weight row read).
// One wave per 16x16 output tile, K-loop in steps of 32 using
// v_wmma_f32_16x16x32_bf16. Block = 128 threads = 4 waves.
// ---------------------------------------------------------------------------
__global__ void __launch_bounds__(128)
wmma_gemm_bias_kernel(const bf16* __restrict__ A1, const bf16* __restrict__ B1, int K1,
                      const bf16* __restrict__ A2, const bf16* __restrict__ B2, int K2,
                      const float* __restrict__ bias, float* __restrict__ C, int N)
{
  const int lane = threadIdx.x & 31;
  const int wave = threadIdx.x >> 5;
  const int tile = blockIdx.x * 4 + wave;
  const int ntn  = N >> 4;
  const int mt   = tile / ntn;
  const int nt   = tile % ntn;
  const int r15  = lane & 15;
  const int grp  = lane >> 4;          // 0: lanes 0-15, 1: lanes 16-31
  const int row  = mt * 16 + r15;      // A-fragment row (M = lane%16)
  const int col  = nt * 16 + r15;      // B-fragment column (N = lane%16)

  v8f acc = {0.f,0.f,0.f,0.f,0.f,0.f,0.f,0.f};

  // ---- contraction 1 ----
  {
    const bf16* arow = A1 + (size_t)row * K1;
    const bf16* bcol = B1 + (size_t)col * K1;
    for (int k0 = 0; k0 < K1; k0 += 32) {
      // A 16-bit 16x32 layout: grp0 holds K {0..7,16..23}, grp1 {8..15,24..31}
      v8bf alo = *(const v8bf*)(arow + k0 + grp * 8);
      v8bf ahi = *(const v8bf*)(arow + k0 + grp * 8 + 16);
      v16bf a;
#pragma unroll
      for (int i = 0; i < 8; i++) { a[i] = alo[i]; a[i + 8] = ahi[i]; }
      // B 32x16 layout: grp0 holds K {0..7,16..23}, grp1 {8..15,24..31} for column = lane%16
      v8bf blo = *(const v8bf*)(bcol + k0 + grp * 8);
      v8bf bhi = *(const v8bf*)(bcol + k0 + grp * 8 + 16);
      v16bf b;
#pragma unroll
      for (int i = 0; i < 8; i++) { b[i] = blo[i]; b[i + 8] = bhi[i]; }
      acc = __builtin_amdgcn_wmma_f32_16x16x32_bf16(false, a, false, b,
                                                    (short)0, acc, false, false);
    }
  }
  // ---- optional contraction 2 ----
  if (K2 > 0) {
    const bf16* arow = A2 + (size_t)row * K2;
    const bf16* bcol = B2 + (size_t)col * K2;
    for (int k0 = 0; k0 < K2; k0 += 32) {
      v8bf alo = *(const v8bf*)(arow + k0 + grp * 8);
      v8bf ahi = *(const v8bf*)(arow + k0 + grp * 8 + 16);
      v16bf a;
#pragma unroll
      for (int i = 0; i < 8; i++) { a[i] = alo[i]; a[i + 8] = ahi[i]; }
      v8bf blo = *(const v8bf*)(bcol + k0 + grp * 8);
      v8bf bhi = *(const v8bf*)(bcol + k0 + grp * 8 + 16);
      v16bf b;
#pragma unroll
      for (int i = 0; i < 8; i++) { b[i] = blo[i]; b[i + 8] = bhi[i]; }
      acc = __builtin_amdgcn_wmma_f32_16x16x32_bf16(false, a, false, b,
                                                    (short)0, acc, false, false);
    }
  }

  // C/D layout: VGPR r -> (M = r + 8*grp, N = lane%16)
  const int m0 = mt * 16 + grp * 8;
  const int n  = nt * 16 + r15;
  const float bv = bias[n];
#pragma unroll
  for (int r = 0; r < 8; r++)
    C[(size_t)(m0 + r) * N + n] = acc[r] + bv;
}

// ---------------------------------------------------------------------------
// Block reduction (blockDim.x == 256)
// ---------------------------------------------------------------------------
__device__ __forceinline__ float block_reduce_sum(float v, float* sdata) {
  const int tid = threadIdx.x;
  sdata[tid] = v;
  __syncthreads();
#pragma unroll
  for (int s = 128; s > 0; s >>= 1) {
    if (tid < s) sdata[tid] += sdata[tid + s];
    __syncthreads();
  }
  float r = sdata[0];
  __syncthreads();
  return r;
}

// ---------------------------------------------------------------------------
// Refinement kernel: one block (256 threads) per batch element.
// Implements LN+relu, S=3 fast-weight refinements with the low-rank history
// representation of A, gating (scan steps) or plain add (last step),
// and history / bf16-activation writeback.
// ---------------------------------------------------------------------------
__global__ void __launch_bounds__(256)
refine_kernel(const float* __restrict__ hbase, float* __restrict__ hist,
              bf16* __restrict__ hbf,
              const float* __restrict__ gamma, const float* __restrict__ beta,
              const float* __restrict__ alpha_ptr, int t, int last)
{
  __shared__ float sdata[256];
  const int b = blockIdx.x;
  const int tid = threadIdx.x;

  // k = 1+softplus(alpha) if alpha>=0 else 1/(1+softplus(-alpha))
  const float alpha = alpha_ptr[0];
  const float kk = (alpha >= 0.f) ? (1.f + log1pf(expf(alpha)))
                                  : (1.f / (1.f + log1pf(expf(-alpha))));

  float hb[4], hv[4], g[4], be[4];
  int jidx[4];
#pragma unroll
  for (int i = 0; i < 4; i++) {
    int j = tid + i * 256;
    jidx[i] = j;
    hb[i] = hbase[(size_t)b * DH + j];
    g[i]  = gamma[j];
    be[i] = beta[j];
  }

  // h = relu(LN(h_base))
  {
    float p = hb[0] + hb[1] + hb[2] + hb[3];
    float mu = block_reduce_sum(p, sdata) * (1.f / DH);
    float vs = 0.f;
#pragma unroll
    for (int i = 0; i < 4; i++) { float d = hb[i] - mu; vs += d * d; }
    float var = block_reduce_sum(vs, sdata) * (1.f / DH);
    float inv = rsqrtf(var + LN_EPS_C);
#pragma unroll
    for (int i = 0; i < 4; i++)
      hv[i] = fmaxf(0.f, (hb[i] - mu) * inv * g[i] + be[i]);
  }

  const int nhist = last ? (T_STEPS - 1) : t;   // #stored h vectors in A
  const int em = nhist - 1;                     // weight = lambda^(em - s)

  for (int rstep = 0; rstep < S_REF; rstep++) {
    // Ah = eta * sum_s lambda^(em-s) (h_s . h) h_s
    float Ah[4] = {0.f, 0.f, 0.f, 0.f};
    for (int s = 0; s < nhist; s++) {
      float hs[4];
      float p = 0.f;
#pragma unroll
      for (int i = 0; i < 4; i++) {
        hs[i] = hist[((size_t)s * BSZ + b) * DH + jidx[i]];
        p += hs[i] * hv[i];
      }
      float d = block_reduce_sum(p, sdata);
      float coef = ETA_C * powf(LAMBDA, (float)(em - s)) * d;
#pragma unroll
      for (int i = 0; i < 4; i++) Ah[i] += coef * hs[i];
    }

    float hnew[4];
    if (!last) {
      float pd = 0.f, p1 = 0.f, p2 = 0.f;
#pragma unroll
      for (int i = 0; i < 4; i++) {
        pd += hv[i] * Ah[i];
        p1 += hv[i] * hv[i];
        p2 += Ah[i] * Ah[i];
      }
      float dotp = block_reduce_sum(pd, sdata);
      float n1 = sqrtf(block_reduce_sum(p1, sdata)) + 1e-6f;
      float n2 = sqrtf(block_reduce_sum(p2, sdata)) + 1e-6f;
      float R = dotp / (n1 * n2 + 1e-6f);
      float Rpos = fminf(fmaxf(R, 0.f), 1.f);
      float a = 1.f - powf(1.f - Rpos, kk);
      float oma2 = 1.f - a * a;
#pragma unroll
      for (int i = 0; i < 4; i++) hnew[i] = oma2 * hb[i] + a * Ah[i];
    } else {
#pragma unroll
      for (int i = 0; i < 4; i++) hnew[i] = hb[i] + Ah[i];
    }

    // h = relu(LN(hnew))
    float p = hnew[0] + hnew[1] + hnew[2] + hnew[3];
    float mu = block_reduce_sum(p, sdata) * (1.f / DH);
    float vs = 0.f;
#pragma unroll
    for (int i = 0; i < 4; i++) { float d = hnew[i] - mu; vs += d * d; }
    float var = block_reduce_sum(vs, sdata) * (1.f / DH);
    float inv = rsqrtf(var + LN_EPS_C);
#pragma unroll
    for (int i = 0; i < 4; i++)
      hv[i] = fmaxf(0.f, (hnew[i] - mu) * inv * g[i] + be[i]);
  }

#pragma unroll
  for (int i = 0; i < 4; i++) {
    if (!last) hist[((size_t)t * BSZ + b) * DH + jidx[i]] = hv[i];
    hbf[(size_t)b * DH + jidx[i]] = (bf16)hv[i];
  }
}

// ---------------------------------------------------------------------------
// Loss: per-batch partials, then mean over batch.
// ---------------------------------------------------------------------------
__global__ void __launch_bounds__(256)
loss_partial_kernel(const float* __restrict__ pred, const float* __restrict__ clean,
                    float* __restrict__ perb)
{
  __shared__ float sdata[256];
  const int b = blockIdx.x;
  const int tid = threadIdx.x;
  float sp = 0.f, sc = 0.f, spc = 0.f, sd = 0.f;
  for (int j = tid; j < DOUT; j += 256) {
    float p = pred[(size_t)b * DOUT + j];
    float c = clean[(size_t)b * DOUT + j];
    sp += p * p; sc += c * c; spc += p * c;
    float d = p - c; sd += d * d;
  }
  float Sp  = block_reduce_sum(sp, sdata);
  float Sc  = block_reduce_sum(sc, sdata);
  float Spc = block_reduce_sum(spc, sdata);
  float Sd  = block_reduce_sum(sd, sdata);
  if (tid == 0) {
    perb[b]       = Sd / (Sc + EPS_C);
    perb[BSZ + b] = Spc / ((sqrtf(Sp) + 1e-6f) * (sqrtf(Sc) + 1e-6f));
  }
}

__global__ void loss_final_kernel(const float* __restrict__ perb,
                                  float* __restrict__ out)
{
  __shared__ float s1[64];
  __shared__ float s2[64];
  const int tid = threadIdx.x;   // 64 threads
  s1[tid] = perb[tid];
  s2[tid] = perb[BSZ + tid];
  __syncthreads();
  for (int s = 32; s > 0; s >>= 1) {
    if (tid < s) { s1[tid] += s1[tid + s]; s2[tid] += s2[tid + s]; }
    __syncthreads();
  }
  if (tid == 0) {
    out[0] = s1[0] * (1.f / BSZ);   // loss
    out[1] = s2[0] * (1.f / BSZ);   // acc
  }
}

// ---------------------------------------------------------------------------
// Host orchestration
// ---------------------------------------------------------------------------
extern "C" void kernel_launch(void* const* d_in, const int* in_sizes, int n_in,
                              void* d_out, int out_size, void* d_ws, size_t ws_size,
                              hipStream_t stream)
{
  const float* z_seq  = (const float*)d_in[0];   // [16,64,512]
  const float* clean  = (const float*)d_in[1];   // [64,512]
  const float* W_h    = (const float*)d_in[2];   // [1024,1024]
  const float* W_g    = (const float*)d_in[3];   // [1024,512]
  const float* b_h    = (const float*)d_in[4];   // [1024]
  const float* gamma  = (const float*)d_in[5];   // [1024]
  const float* beta   = (const float*)d_in[6];   // [1024]
  const float* alpha  = (const float*)d_in[7];   // [1]
  const float* head_W = (const float*)d_in[8];   // [512,1024]
  const float* head_b = (const float*)d_in[9];   // [512]
  float* out = (float*)d_out;                    // [loss, acc]

  char* ws = (char*)d_ws;
  size_t off = 0;
  auto alloc = [&](size_t bytes) -> char* {
    char* p = ws + off;
    off += (bytes + 255) & ~(size_t)255;
    return p;
  };

  bf16* Whb  = (bf16*)alloc((size_t)DH * DH * sizeof(bf16));
  bf16* Wgb  = (bf16*)alloc((size_t)DH * DG * sizeof(bf16));
  bf16* Hwb  = (bf16*)alloc((size_t)DOUT * DH * sizeof(bf16));
  bf16* Zb   = (bf16*)alloc((size_t)T_STEPS * BSZ * DG * sizeof(bf16));
  float* hist = (float*)alloc((size_t)(T_STEPS - 1) * BSZ * DH * sizeof(float));
  bf16* hbf  = (bf16*)alloc((size_t)BSZ * DH * sizeof(bf16));
  float* hbase = (float*)alloc((size_t)BSZ * DH * sizeof(float));
  float* pred  = (float*)alloc((size_t)BSZ * DOUT * sizeof(float));
  float* perb  = (float*)alloc((size_t)2 * BSZ * sizeof(float));

  // One-time precision conversion of constants + inputs (L2-resident afterwards)
  convert_f32_bf16_kernel<<<(DH * DH + 255) / 256, 256, 0, stream>>>(W_h, Whb, DH * DH);
  convert_f32_bf16_kernel<<<(DH * DG + 255) / 256, 256, 0, stream>>>(W_g, Wgb, DH * DG);
  convert_f32_bf16_kernel<<<(DOUT * DH + 255) / 256, 256, 0, stream>>>(head_W, Hwb, DOUT * DH);
  convert_f32_bf16_kernel<<<(T_STEPS * BSZ * DG + 255) / 256, 256, 0, stream>>>(
      z_seq, Zb, T_STEPS * BSZ * DG);
  zero_bf16_kernel<<<(BSZ * DH + 255) / 256, 256, 0, stream>>>(hbf, BSZ * DH);

  // Recurrent scan: GEMM (WMMA bf16) + refinement per timestep
  for (int t = 0; t < T_STEPS; t++) {
    // h_base = h @ W_h^T + z_t @ W_g^T + b_h   -> 256 tiles, 4 waves/block
    wmma_gemm_bias_kernel<<<64, 128, 0, stream>>>(
        hbf, Whb, DH,
        Zb + (size_t)t * BSZ * DG, Wgb, DG,
        b_h, hbase, DH);
    refine_kernel<<<BSZ, 256, 0, stream>>>(
        hbase, hist, hbf, gamma, beta, alpha, t, (t == T_STEPS - 1) ? 1 : 0);
  }

  // pred = h @ head_W^T + head_b   -> 128 tiles, 4 waves/block
  wmma_gemm_bias_kernel<<<32, 128, 0, stream>>>(
      hbf, Hwb, DH, (const bf16*)nullptr, (const bf16*)nullptr, 0,
      head_b, pred, DOUT);

  loss_partial_kernel<<<BSZ, 256, 0, stream>>>(pred, clean, perb);
  loss_final_kernel<<<1, 64, 0, stream>>>(perb, out);
}
